// LowRankAttention_5222680232430
// MI455X (gfx1250) — compile-verified, matrix operand order checked by reading, not run on hardware
//
#include <hip/hip_runtime.h>
#include <hip/hip_bf16.h>

// ---------------------------------------------------------------------------
// Low-rank attention, MI455X (gfx1250, wave32, WMMA + TDM).
//   K1: qkv = x@Wqkv+b -> V (bf16, tile-packed for WMMA-B), Q_lr/K_lr
//       (bf16 [B,H,N,32], rank 8 zero-padded to 32, scale folded into Q)
//   K2: flash attention, K/V tiles streamed by the Tensor Data Mover
//       (tensor_load_to_lds + s_wait_tensorcnt, double buffered)
//   K3: out = Ows@Wproj + b
// GEMMs: bf16 in, f32 accumulate, v_wmma_f32_16x16x32_bf16; GEMM kernels are
// register-blocked M=32/wave so each B fragment feeds 2 WMMAs.
// ---------------------------------------------------------------------------

#define DEVI __device__ __forceinline__

typedef __attribute__((ext_vector_type(16))) __bf16        v16bf;
typedef __attribute__((ext_vector_type(8)))  float         v8f;
typedef __attribute__((ext_vector_type(4)))  unsigned int  v4u;
typedef __attribute__((ext_vector_type(4)))  unsigned int  u32x4;
typedef __attribute__((ext_vector_type(8)))  int           i32x8;
typedef __attribute__((ext_vector_type(4)))  int           i32x4;

constexpr int Bc = 4, Nc = 2048, Cc = 768, Hc = 12, Dc = 64, Rc = 8;
constexpr int LRP = 32;                               // padded rank (WMMA K)
constexpr float SCALE = 0.35355339059327373f;         // R^-0.5

#if __has_builtin(__builtin_amdgcn_tensor_load_to_lds)
#define HAVE_TDM 1
#else
#define HAVE_TDM 0
#endif

// D# group1 dword0: data_size=2B; optional LDS pad 16B after every 64B
// (reproduces the [64][40] bf16 bank-conflict-free layout).
constexpr unsigned TDM_FLAGS_PAD = (1u << 16) | (1u << 20) | (3u << 22) | (3u << 25);
constexpr unsigned TDM_FLAGS_LIN = (1u << 16);

DEVI v8f v8f_zero() { v8f z = {0.f,0.f,0.f,0.f,0.f,0.f,0.f,0.f}; return z; }

DEVI v8f wmma_bf16(v16bf a, v16bf b, v8f c) {
  return __builtin_amdgcn_wmma_f32_16x16x32_bf16(false, a, false, b,
                                                 (short)0, c, false, false);
}

// A-fragment (16-bit A 16x32, wave32): lane half*16+m; the 8 K-pairs form two
// contiguous 16-byte runs -> two ds_load_b128. lda (elems) * 2 must be 16B mult.
DEVI v16bf frag_A(const __bf16* base, int lda, int m, int half) {
  union { v16bf v; v4u q[2]; } f;
  const __bf16* p = base + m * lda + half * 8;
  f.q[0] = *(const v4u*)(p);
  f.q[1] = *(const v4u*)(p + 16);
  return f.v;
}

// B-fragment from pair-packed LDS: element (k,n) lives at [(k>>1)*ldp + n*2 +
// (k&1)]; each VGPR = one aligned dword load.
DEVI v16bf frag_Bpk(const __bf16* base, int ldp, int n, int half) {
  union { v16bf v; unsigned int d[8]; } f;
#pragma unroll
  for (int i = 0; i < 8; ++i) {
    int kb = ((i < 4) ? (2 * i) : (2 * i + 8)) + half * 8;
    f.d[i] = *(const unsigned int*)(base + (kb >> 1) * ldp + n * 2);
  }
  return f.v;
}

DEVI unsigned lds_addr_of(const void* p) {           // LDS aperture: low 32 bits
  return (unsigned)(unsigned long long)p;
}

DEVI void tdm_wait0() {
#if __has_builtin(__builtin_amdgcn_s_wait_tensorcnt)
  __builtin_amdgcn_s_wait_tensorcnt(0);
#else
  asm volatile("s_wait_tensorcnt 0x0" ::: "memory");
#endif
}

// One 1-D TDM copy: nelem bf16 elements from global -> LDS (optional padding).
// 6-arg toolchain form: (g0, g1, g2, g3, g4, cpol); groups 2..4 unused (1-D).
DEVI void tdm_load_1d(unsigned lds, const void* gaddr, unsigned nelem,
                      unsigned flags) {
#if HAVE_TDM
  unsigned long long ga = (unsigned long long)gaddr;
  u32x4 g0;
  g0[0] = 1u;                                        // count=1 user descriptor
  g0[1] = lds;                                       // lds_addr
  g0[2] = (unsigned)(ga & 0xffffffffu);              // global_addr[31:0]
  g0[3] = (unsigned)((ga >> 32) & 0x01ffffffu) | (2u << 30);  // [56:32]+type=2
  i32x8 g1 = {0, 0, 0, 0, 0, 0, 0, 0};
  g1[0] = (int)flags;
  g1[1] = (int)(nelem << 16);                        // tensor_dim0[15:0]
  g1[2] = (int)((nelem >> 16) | (1u << 16));         // tensor_dim0 hi, dim1=1
  g1[3] = (int)(nelem << 16);                        // tile_dim0 (<=65535)
  i32x4 gz4 = {0, 0, 0, 0};
  i32x8 gz8 = {0, 0, 0, 0, 0, 0, 0, 0};
  __builtin_amdgcn_tensor_load_to_lds(g0, g1, gz4, gz4, gz8, 0);
#else
  (void)lds; (void)gaddr; (void)nelem; (void)flags;
#endif
}

// ---------------------------------------------------------------------------
// Kernel 1: QKV GEMM + low-rank projection.
// grid = (B*N/128, H, 3[q,k,v]), block = 128 (4 waves), tile 128x64 (M=32 per
// wave, B fragments reused 2x), K-step 32.
// ---------------------------------------------------------------------------
__global__ __launch_bounds__(128) void qkv_lowrank_kernel(
    const float* __restrict__ x, const float* __restrict__ Wqkv,
    const float* __restrict__ bqkv, const float* __restrict__ P,
    __bf16* __restrict__ Vws, __bf16* __restrict__ Qlr,
    __bf16* __restrict__ Klr) {
  __shared__ __align__(16) __bf16 As[128 * 40];  // x tile, 80B rows
  __shared__ __align__(16) __bf16 Bs[16 * 128];  // W tile, pair-packed
  __shared__ float Cs[128 * 65];
  __shared__ float Psh[Dc * Rc];

  const int tid  = threadIdx.x;
  const int wave = tid >> 5, lane = tid & 31, half = lane >> 4, ml = lane & 15;
  const int h = blockIdx.y, z = blockIdx.z;
  const int colBase = z * Cc + h * Dc;
  const int rowBase = blockIdx.x * 128;

  if (z < 2)
    for (int i = tid; i < Dc * Rc; i += 128) Psh[i] = P[h * Dc * Rc + i];

  v8f acc[2][4];
#pragma unroll
  for (int mm = 0; mm < 2; ++mm)
#pragma unroll
    for (int t = 0; t < 4; ++t) acc[mm][t] = v8f_zero();

  for (int kt = 0; kt < Cc; kt += 32) {
    __syncthreads();
    {  // A tile: one full 32-elem row per thread -> 4x ds_store_b128
      const float* src = x + (size_t)(rowBase + tid) * Cc + kt;
      union { v16bf v; v4u q[2]; } lo, hi;
#pragma unroll
      for (int e = 0; e < 16; ++e) {
        lo.v[e] = (__bf16)src[e];
        hi.v[e] = (__bf16)src[16 + e];
      }
      *(v4u*)(&As[tid * 40 + 0])  = lo.q[0];
      *(v4u*)(&As[tid * 40 + 8])  = lo.q[1];
      *(v4u*)(&As[tid * 40 + 16]) = hi.q[0];
      *(v4u*)(&As[tid * 40 + 24]) = hi.q[1];
    }
    {  // B tile: row pairs packed into dwords
#pragma unroll
      for (int p = tid; p < 16 * 64; p += 128) {
        int kp = p >> 6, n = p & 63;
        const float* w = Wqkv + (size_t)(kt + 2 * kp) * (3 * Cc) + colBase + n;
        union { unsigned u; __bf16 hh[2]; } pr;
        pr.hh[0] = (__bf16)w[0];
        pr.hh[1] = (__bf16)w[3 * Cc];
        *(unsigned*)(&Bs[kp * 128 + n * 2]) = pr.u;
      }
    }
    __syncthreads();
    v16bf a0 = frag_A(&As[(wave * 32) * 40],      40, ml, half);
    v16bf a1 = frag_A(&As[(wave * 32 + 16) * 40], 40, ml, half);
#pragma unroll
    for (int t = 0; t < 4; ++t) {
      v16bf b = frag_Bpk(Bs, 128, t * 16 + ml, half);
      acc[0][t] = wmma_bf16(a0, b, acc[0][t]);
      acc[1][t] = wmma_bf16(a1, b, acc[1][t]);
    }
  }

  if (z == 2) {  // V: store pair-packed per 64-key tile (TDM-linear in K2)
#pragma unroll
    for (int mm = 0; mm < 2; ++mm)
#pragma unroll
      for (int t = 0; t < 4; ++t) {
        int col = t * 16 + ml;
        float bias = bqkv[colBase + col];
#pragma unroll
        for (int r = 0; r < 8; ++r) {
          int gr = rowBase + wave * 32 + mm * 16 + half * 8 + r;
          int bb = gr >> 11, n = gr & (Nc - 1);
          int jt = n >> 6, kk = n & 63;
          size_t idx = ((size_t)(bb * Hc + h) * 32 + jt) * 4096 +
                       (size_t)(kk >> 1) * 128 + col * 2 + (kk & 1);
          Vws[idx] = (__bf16)(acc[mm][t][r] + bias);
        }
      }
  } else {       // Q/K: tile @ P_h -> rank-32 zero-padded
#pragma unroll
    for (int mm = 0; mm < 2; ++mm)
#pragma unroll
      for (int t = 0; t < 4; ++t) {
        int col = t * 16 + ml;
        float bias = bqkv[colBase + col];
#pragma unroll
        for (int r = 0; r < 8; ++r)
          Cs[(wave * 32 + mm * 16 + half * 8 + r) * 65 + col] =
              acc[mm][t][r] + bias;
      }
    __syncthreads();
    __bf16* dst = (z == 0) ? Qlr : Klr;
    float sc = (z == 0) ? SCALE : 1.0f;
    for (int i = tid; i < 128 * Rc; i += 128) {
      int r = i >> 3, rr = i & 7;
      float s = 0.f;
#pragma unroll
      for (int d = 0; d < Dc; ++d) s += Cs[r * 65 + d] * Psh[d * Rc + rr];
      int gr = rowBase + r;
      int bb = gr >> 11, n = gr & (Nc - 1);
      dst[((size_t)(bb * Hc + h) * Nc + n) * LRP + rr] = (__bf16)(s * sc);
    }
    for (int i = tid; i < 128 * (LRP - Rc); i += 128) {
      int r = i / 24, pc = Rc + i % 24;
      int gr = rowBase + r;
      int bb = gr >> 11, n = gr & (Nc - 1);
      dst[((size_t)(bb * Hc + h) * Nc + n) * LRP + pc] = (__bf16)0.0f;
    }
  }
}

// ---------------------------------------------------------------------------
// Kernel 2: flash attention over rank-8 logits, TDM-streamed K/V tiles.
// grid = (N/64 query tiles, B*H), block = 128 (4 waves x 16 query rows).
// ---------------------------------------------------------------------------
__global__ __launch_bounds__(128) void lowrank_flash_attn_kernel(
    const __bf16* __restrict__ Qlr, const __bf16* __restrict__ Klr,
    const __bf16* __restrict__ Vws, float* __restrict__ Ows) {
  __shared__ __align__(16) __bf16 Qs[64 * 40];       // Q_lr tile (padded rows)
  __shared__ __align__(16) __bf16 Ks[2][64 * 40];    // K_lr, double buffered
  __shared__ __align__(16) __bf16 Vs[2][32 * 128];   // V pair-packed, dbl buf
  __shared__ __align__(16) __bf16 Ps[4 * 16 * 72];   // per-wave P tile

  const int tid = threadIdx.x;
  const int wave = tid >> 5, lane = tid & 31, half = lane >> 4, ml = lane & 15;
  const int q0 = blockIdx.x * 64;
  const int bh = blockIdx.y;
  const int b = bh / Hc, h = bh % Hc;

  const __bf16* Qg = Qlr + (size_t)(bh * Nc + q0) * LRP;   // 64x32 contiguous
  const __bf16* Kg = Klr + (size_t)bh * Nc * LRP;
  const __bf16* Vg = Vws + (size_t)bh * 32 * 4096;         // 4096/tile

#if HAVE_TDM
  if (wave == 0) {
    tdm_load_1d(lds_addr_of(Qs),    Qg, 64 * LRP, TDM_FLAGS_PAD);
    tdm_load_1d(lds_addr_of(Ks[0]), Kg, 64 * LRP, TDM_FLAGS_PAD);
    tdm_load_1d(lds_addr_of(Vs[0]), Vg, 4096,     TDM_FLAGS_LIN);
    tdm_wait0();
  }
  __syncthreads();
#else
  for (int i = tid; i < 64 * LRP; i += 128) {
    int r = i >> 5, k = i & 31;
    Qs[r * 40 + k] = Qg[(size_t)r * LRP + k];
  }
  __syncthreads();
#endif
  v16bf aq = frag_A(&Qs[wave * 16 * 40], 40, ml, half);

  v8f acc[4];
  float mrow[8], lrow[8];
#pragma unroll
  for (int t = 0; t < 4; ++t) acc[t] = v8f_zero();
#pragma unroll
  for (int r = 0; r < 8; ++r) { mrow[r] = -1e30f; lrow[r] = 0.f; }

  for (int j = 0; j < Nc / 64; ++j) {
    const int cur = j & 1;
#if HAVE_TDM
    if (wave == 0 && j + 1 < Nc / 64) {   // prefetch next tile, other buffer
      tdm_load_1d(lds_addr_of(Ks[cur ^ 1]), Kg + (size_t)(j + 1) * 64 * LRP,
                  64 * LRP, TDM_FLAGS_PAD);
      tdm_load_1d(lds_addr_of(Vs[cur ^ 1]), Vg + (size_t)(j + 1) * 4096,
                  4096, TDM_FLAGS_LIN);
    }
#else
    __syncthreads();
    for (int i = tid; i < 64 * LRP; i += 128) {
      int r = i >> 5, k = i & 31;
      Ks[cur][r * 40 + k] = Kg[(size_t)(j * 64 + r) * LRP + k];
    }
    for (int i = tid; i < 4096; i += 128) Vs[cur][i] = Vg[(size_t)j * 4096 + i];
    __syncthreads();
#endif

    // S[16x64] = Q_lr @ K_lr^T ; B(k=r,n=key) == Ks[key][r] -> frag_A on Ks
    v8f s[4];
#pragma unroll
    for (int t = 0; t < 4; ++t) {
      v16bf bk = frag_A(&Ks[cur][(t * 16) * 40], 40, ml, half);
      s[t] = wmma_bf16(aq, bk, v8f_zero());
    }

    // online softmax: row = wave*16 + half*8 + r; cols in 16-lane group
#pragma unroll
    for (int r = 0; r < 8; ++r) {
      float tmax = fmaxf(fmaxf(s[0][r], s[1][r]), fmaxf(s[2][r], s[3][r]));
#pragma unroll
      for (int off = 1; off < 16; off <<= 1)
        tmax = fmaxf(tmax, __shfl_xor(tmax, off, 32));
      float mnew = fmaxf(mrow[r], tmax);
      float corr = __expf(mrow[r] - mnew);
      float rsum = 0.f;
#pragma unroll
      for (int t = 0; t < 4; ++t) {
        float p = __expf(s[t][r] - mnew);
        s[t][r] = p;
        rsum += p;
      }
#pragma unroll
      for (int off = 1; off < 16; off <<= 1)
        rsum += __shfl_xor(rsum, off, 32);
      lrow[r] = lrow[r] * corr + rsum;
      mrow[r] = mnew;
#pragma unroll
      for (int t = 0; t < 4; ++t) acc[t][r] *= corr;
    }

    // restage P (C-layout -> A-layout) through LDS
#pragma unroll
    for (int t = 0; t < 4; ++t)
#pragma unroll
      for (int r = 0; r < 8; ++r)
        Ps[(wave * 16 + half * 8 + r) * 72 + t * 16 + ml] = (__bf16)s[t][r];
    __syncthreads();

    // O[16x64] += P @ V ; two K=32 steps, V read pair-packed
#pragma unroll
    for (int ks = 0; ks < 2; ++ks) {
      v16bf ap = frag_A(&Ps[wave * 16 * 72 + ks * 32], 72, ml, half);
#pragma unroll
      for (int t = 0; t < 4; ++t) {
        v16bf bv = frag_Bpk(&Vs[cur][ks * 16 * 128], 128, t * 16 + ml, half);
        acc[t] = wmma_bf16(ap, bv, acc[t]);
      }
    }
#if HAVE_TDM
    if (wave == 0 && j + 1 < Nc / 64) tdm_wait0();
    __syncthreads();
#endif
  }

#pragma unroll
  for (int r = 0; r < 8; ++r) {
    float inv = 1.0f / lrow[r];
    int row = q0 + wave * 16 + half * 8 + r;
#pragma unroll
    for (int t = 0; t < 4; ++t)
      Ows[((size_t)b * Nc + row) * Cc + h * Dc + t * 16 + ml] = acc[t][r] * inv;
  }
}

// ---------------------------------------------------------------------------
// Kernel 3: output projection. grid = (B*N/128, C/64), block = 128,
// tile 128x64 (M=32 per wave).
// ---------------------------------------------------------------------------
__global__ __launch_bounds__(128) void proj_kernel(
    const float* __restrict__ Ows, const float* __restrict__ Wp,
    const float* __restrict__ bp, float* __restrict__ out) {
  __shared__ __align__(16) __bf16 As[128 * 40];
  __shared__ __align__(16) __bf16 Bs[16 * 128];

  const int tid = threadIdx.x;
  const int wave = tid >> 5, lane = tid & 31, half = lane >> 4, ml = lane & 15;
  const int rowBase = blockIdx.x * 128;
  const int colBase = blockIdx.y * 64;

  v8f acc[2][4];
#pragma unroll
  for (int mm = 0; mm < 2; ++mm)
#pragma unroll
    for (int t = 0; t < 4; ++t) acc[mm][t] = v8f_zero();

  for (int kt = 0; kt < Cc; kt += 32) {
    __syncthreads();
    {
      const float* src = Ows + (size_t)(rowBase + tid) * Cc + kt;
      union { v16bf v; v4u q[2]; } lo, hi;
#pragma unroll
      for (int e = 0; e < 16; ++e) {
        lo.v[e] = (__bf16)src[e];
        hi.v[e] = (__bf16)src[16 + e];
      }
      *(v4u*)(&As[tid * 40 + 0])  = lo.q[0];
      *(v4u*)(&As[tid * 40 + 8])  = lo.q[1];
      *(v4u*)(&As[tid * 40 + 16]) = hi.q[0];
      *(v4u*)(&As[tid * 40 + 24]) = hi.q[1];
    }
#pragma unroll
    for (int p = tid; p < 16 * 64; p += 128) {
      int kp = p >> 6, n = p & 63;
      const float* w = Wp + (size_t)(kt + 2 * kp) * Cc + colBase + n;
      union { unsigned u; __bf16 hh[2]; } pr;
      pr.hh[0] = (__bf16)w[0];
      pr.hh[1] = (__bf16)w[Cc];
      *(unsigned*)(&Bs[kp * 128 + n * 2]) = pr.u;
    }
    __syncthreads();
    v16bf a0 = frag_A(&As[(wave * 32) * 40],      40, ml, half);
    v16bf a1 = frag_A(&As[(wave * 32 + 16) * 40], 40, ml, half);
#pragma unroll
    for (int t = 0; t < 4; ++t) {
      v16bf b = frag_Bpk(Bs, 128, t * 16 + ml, half);
      acc[0][t] = wmma_bf16(a0, b, acc[0][t]);
      acc[1][t] = wmma_bf16(a1, b, acc[1][t]);
    }
  }

#pragma unroll
  for (int mm = 0; mm < 2; ++mm)
#pragma unroll
    for (int t = 0; t < 4; ++t) {
      int col = colBase + t * 16 + ml;
      float bias = bp[col];
#pragma unroll
      for (int r = 0; r < 8; ++r) {
        int gr = rowBase + wave * 32 + mm * 16 + half * 8 + r;
        out[(size_t)gr * Cc + col] = acc[mm][t][r] + bias;
      }
    }
}

// ---------------------------------------------------------------------------
extern "C" void kernel_launch(void* const* d_in, const int* in_sizes, int n_in,
                              void* d_out, int out_size, void* d_ws,
                              size_t ws_size, hipStream_t stream) {
  (void)in_sizes; (void)n_in; (void)out_size; (void)ws_size;
  const float* x    = (const float*)d_in[0];
  const float* Wqkv = (const float*)d_in[1];
  const float* bqkv = (const float*)d_in[2];
  const float* P    = (const float*)d_in[3];
  const float* Wp   = (const float*)d_in[4];
  const float* bp   = (const float*)d_in[5];
  float* out = (float*)d_out;

  // workspace layout (48 MiB total)
  char* ws = (char*)d_ws;
  __bf16* Vws = (__bf16*)(ws);             // B*H*32 tiles * 4096 bf16 = 12 MiB
  __bf16* Qlr = (__bf16*)(ws + 12582912);  // B*H*N*32 bf16 = 6 MiB
  __bf16* Klr = (__bf16*)(ws + 18874368);  // B*H*N*32 bf16 = 6 MiB
  float*  Ows = (float*)(ws + 25165824);   // B*N*C f32 = 24 MiB

  qkv_lowrank_kernel<<<dim3(Bc * Nc / 128, Hc, 3), 128, 0, stream>>>(
      x, Wqkv, bqkv, P, Vws, Qlr, Klr);
  lowrank_flash_attn_kernel<<<dim3(Nc / 64, Bc * Hc), 128, 0, stream>>>(
      Qlr, Klr, Vws, Ows);
  proj_kernel<<<dim3(Bc * Nc / 128, Cc / 64), 128, 0, stream>>>(
      Ows, Wp, bp, out);
}
